// DynamicConv2d_3796751089749
// MI455X (gfx1250) — matrix operand order repaired and checked by reading, not compiled, not run on hardware
//
#include <hip/hip_runtime.h>
#include <hip/hip_bf16.h>

// CDNA5 / gfx1250 wave32 WMMA types
typedef __attribute__((ext_vector_type(16))) __bf16 v16bf;
typedef __attribute__((ext_vector_type(8)))  float  v8f;

// Problem constants (from reference)
#define BB 32
#define CC 256
#define HH 64
#define WW 64
#define EE 4
#define OO 256
#define KK 2304      // 9 taps * 256 channels   (weight layout: [o][tap][c])
#define PP 4096      // H*W

__device__ __forceinline__ unsigned short f2bf(float f) {
    union { float f; unsigned u; } cv; cv.f = f;
    unsigned u = cv.u;
    u += 0x7FFFu + ((u >> 16) & 1u);   // round-to-nearest-even
    return (unsigned short)(u >> 16);
}

// ---------------------------------------------------------------------------
// Kernel 1: fused global-average-pool (mean over H*W) + f32 -> bf16 convert.
// grid = B*C blocks, 256 threads; each thread handles 16 contiguous floats.
// ---------------------------------------------------------------------------
__global__ void pool_convert_kernel(const float* __restrict__ x,
                                    float* __restrict__ pooled,
                                    unsigned short* __restrict__ xbf) {
    const int bc = blockIdx.x;
    const int t  = threadIdx.x;
    const size_t base = ((size_t)bc << 12) + ((size_t)t << 4);
    const float4* s4 = reinterpret_cast<const float4*>(x + base);
    float4 f[4];
    float sum = 0.f;
    unsigned pk[8];
#pragma unroll
    for (int i = 0; i < 4; ++i) {
        f[i] = s4[i];
        sum += f[i].x + f[i].y + f[i].z + f[i].w;
        pk[2 * i]     = (unsigned)f2bf(f[i].x) | ((unsigned)f2bf(f[i].y) << 16);
        pk[2 * i + 1] = (unsigned)f2bf(f[i].z) | ((unsigned)f2bf(f[i].w) << 16);
    }
    uint4* dst = reinterpret_cast<uint4*>(xbf + base);
    dst[0] = make_uint4(pk[0], pk[1], pk[2], pk[3]);
    dst[1] = make_uint4(pk[4], pk[5], pk[6], pk[7]);

    __shared__ float sm[256];
    sm[t] = sum;
    __syncthreads();
#pragma unroll
    for (int s = 128; s > 0; s >>= 1) {
        if (t < s) sm[t] += sm[t + s];
        __syncthreads();
    }
    if (t == 0) pooled[bc] = sm[0] * (1.0f / 4096.0f);
}

// ---------------------------------------------------------------------------
// Kernel 2: router logits + softmax. One wave; thread b handles sample b.
// ---------------------------------------------------------------------------
__global__ void router_kernel(const float* __restrict__ pooled,
                              const float* __restrict__ rw,
                              const float* __restrict__ rb,
                              float* __restrict__ attn) {
    const int b = threadIdx.x;
    if (b >= BB) return;
    float lg[EE];
#pragma unroll
    for (int e = 0; e < EE; ++e) {
        float s = rb[e];
        const float* pb = pooled + b * CC;
        const float* we = rw + e * CC;
        for (int c = 0; c < CC; ++c) s += pb[c] * we[c];
        lg[e] = s;
    }
    float mx = lg[0];
#pragma unroll
    for (int e = 1; e < EE; ++e) mx = fmaxf(mx, lg[e]);
    float den = 0.f;
#pragma unroll
    for (int e = 0; e < EE; ++e) { lg[e] = __expf(lg[e] - mx); den += lg[e]; }
    const float inv = 1.0f / den;
#pragma unroll
    for (int e = 0; e < EE; ++e) attn[b * EE + e] = lg[e] * inv;
}

// ---------------------------------------------------------------------------
// Kernel 3: combine expert weights per sample -> bf16 (CondConv identity),
// writing the TAP-MAJOR layout  wb[b][o][tap][c]  (tap = 3*di+dj).
//   wb[b][o][r*256+c] = sum_e attn[b][e] * expert_w[e][o][c][r]
// 2 destination elements per thread (consecutive c, packed into one dword).
// ---------------------------------------------------------------------------
__global__ void combine_kernel(const float* __restrict__ attn,
                               const float* __restrict__ ew,
                               unsigned int* __restrict__ wbp) {
    const long gid = (long)blockIdx.x * blockDim.x + threadIdx.x;  // < B*O*KK/2
    const long OK  = (long)OO * KK;                                 // 589824
    const long idx = gid * 2;                 // destination linear index
    const int  b   = (int)(idx / OK);
    const long rem = idx - (long)b * OK;
    const int  o   = (int)(rem / KK);
    const int  t2  = (int)(rem - (long)o * KK);
    const int  r   = t2 >> 8;                 // tap 0..8
    const int  c   = t2 & 255;                // channel (even)
    const long src0 = (long)o * KK + (long)c * 9 + r;   // ew layout [o][c][r]
    const long src1 = src0 + 9;
    const float a0 = attn[b * EE + 0];
    const float a1 = attn[b * EE + 1];
    const float a2 = attn[b * EE + 2];
    const float a3 = attn[b * EE + 3];
    float w0 = a0 * ew[src0]          + a1 * ew[OK + src0]
             + a2 * ew[2 * OK + src0] + a3 * ew[3 * OK + src0];
    float w1 = a0 * ew[src1]          + a1 * ew[OK + src1]
             + a2 * ew[2 * OK + src1] + a3 * ew[3 * OK + src1];
    wbp[gid] = (unsigned)f2bf(w0) | ((unsigned)f2bf(w1) << 16);
}

// ---------------------------------------------------------------------------
// Kernel 4: per-sample implicit GEMM, bf16 WMMA, f32 accumulate.
//   Y_b[O=256, P=4096] = W_b[256, 2304] * im2col(x_b)[2304, 4096]
// K dimension ordered [tap][c]: kt-th 32-chunk is tap = kt>>3, c0 = (kt&7)*32.
// B-stage gather per thread: ONE bounds check + 16 channel-plane-strided
// bf16 loads, packed and committed with two ds_store_b128 (no div/mod).
// grid = (P/128, O/128, B); 256 threads = 8 waves (4 along M, 2 along N).
// ---------------------------------------------------------------------------
__global__ __launch_bounds__(256, 2)
void condconv_gemm_kernel(const unsigned short* __restrict__ xbf,
                          const unsigned short* __restrict__ wb,
                          float* __restrict__ out) {
    constexpr int LDA = 40;       // ushort row stride (pad, keeps 16B align)
    constexpr int LDB = 40;
    constexpr int NKT = KK / 32;  // 72 = 9 taps * 8 channel-chunks

    __shared__ __align__(16) unsigned short As[2][128 * LDA];
    __shared__ __align__(16) unsigned short Bs[2][128 * LDB];

    const int b  = blockIdx.z;
    const int o0 = blockIdx.y * 128;
    const int p0 = blockIdx.x * 128;
    const int t    = threadIdx.x;
    const int lane = t & 31;
    const int wave = t >> 5;
    const int wm   = wave & 3;   // M sub-tile (4 x 32 rows)
    const int wn   = wave >> 2;  // N sub-tile (2 x 64 cols)
    const int half = lane >> 4;  // lane group (0: lanes 0-15, 1: 16-31)
    const int lm   = lane & 15;

    const unsigned short* wbb = wb  + (size_t)b * OO * KK;
    const unsigned short* xb  = xbf + (size_t)b * CC * PP;

    // staging roles: thread -> (row, 16-k half)
    const int a_row = t >> 1, a_kh = t & 1;
    const int b_prw = t >> 1, b_kh = t & 1;
    const int p  = p0 + b_prw;
    const int ph = p >> 6, pw = p & 63;

    v8f acc[2][4] = {};

    auto stage = [&](int kt, int buf) {
        // --- A tile: 128 x 32 bf16, straight copy ([tap][c] flat == kt*32) ---
        const uint4* ga = reinterpret_cast<const uint4*>(
            wbb + (size_t)(o0 + a_row) * KK + kt * 32 + a_kh * 16);
        uint4 va0 = ga[0], va1 = ga[1];
        uint4* sa = reinterpret_cast<uint4*>(&As[buf][a_row * LDA + a_kh * 16]);
        sa[0] = va0; sa[1] = va1;

        // --- B tile: tap-shifted channel gather, transposed store [p][k] ---
        const int tap = kt >> 3;            // uniform across block
        const int c0  = (kt & 7) * 32;
        const int di  = tap / 3;            // uniform -> SALU
        const int dj  = tap - di * 3;
        const int ih  = ph + di - 1;
        const int iw  = pw + dj - 1;
        const bool valid = ((unsigned)ih < 64u) && ((unsigned)iw < 64u);

        unsigned short vv[16];
        if (valid) {
            const unsigned short* basep =
                xb + (size_t)(c0 + b_kh * 16) * PP + (ih << 6) + iw;
#pragma unroll
            for (int i = 0; i < 16; ++i) vv[i] = basep[(size_t)i * PP];
        } else {
#pragma unroll
            for (int i = 0; i < 16; ++i) vv[i] = 0;
        }
        unsigned pk[8];
#pragma unroll
        for (int i = 0; i < 8; ++i)
            pk[i] = (unsigned)vv[2 * i] | ((unsigned)vv[2 * i + 1] << 16);
        uint4* sb = reinterpret_cast<uint4*>(&Bs[buf][b_prw * LDB + b_kh * 16]);
        sb[0] = make_uint4(pk[0], pk[1], pk[2], pk[3]);
        sb[1] = make_uint4(pk[4], pk[5], pk[6], pk[7]);
    };

    stage(0, 0);
    __syncthreads();

    for (int kt = 0; kt < NKT; ++kt) {
        const int buf = kt & 1;
        if (kt + 1 < NKT) stage(kt + 1, buf ^ 1);

        union V16 { uint4 q[2]; v16bf v; };
        V16 afr[2], bfr[4];
        // A fragment (16x32): lanes 0-15 K=0..15, lanes 16-31 K=16..31
        // (contiguous, matches B). Two b128 LDS loads.
#pragma unroll
        for (int mi = 0; mi < 2; ++mi) {
            const int row = wm * 32 + mi * 16 + lm;
            const uint4* pa = reinterpret_cast<const uint4*>(&As[buf][row * LDA]);
            afr[mi].q[0] = pa[2 * half];      // K = half*16 .. +7
            afr[mi].q[1] = pa[2 * half + 1];  // K = half*16 + 8 .. +15
        }
        // B fragment (32x16): lane n = lm, K range = half*16..+15, contiguous
        // in the transposed LDS layout -> two b128 loads.
#pragma unroll
        for (int ni = 0; ni < 4; ++ni) {
            const int col = wn * 64 + ni * 16 + lm;
            const uint4* pb = reinterpret_cast<const uint4*>(
                &Bs[buf][col * LDB + half * 16]);
            bfr[ni].q[0] = pb[0];
            bfr[ni].q[1] = pb[1];
        }
#pragma unroll
        for (int mi = 0; mi < 2; ++mi)
#pragma unroll
            for (int ni = 0; ni < 4; ++ni)
                acc[mi][ni] = __builtin_amdgcn_wmma_f32_16x16x32_bf16(
                    false, afr[mi].v, false, bfr[ni].v,
                    (short)0, acc[mi][ni], false, false);
        __syncthreads();
    }

    // C/D layout: VGPR v, lanes 0-15 -> M=v, lanes 16-31 -> M=v+8; N=lane%16.
#pragma unroll
    for (int mi = 0; mi < 2; ++mi) {
#pragma unroll
        for (int ni = 0; ni < 4; ++ni) {
            const int oo = o0 + wm * 32 + mi * 16 + half * 8;
            const int pp = p0 + wn * 64 + ni * 16 + lm;
            float* dst = out + ((size_t)b * OO + oo) * PP + pp;
#pragma unroll
            for (int v = 0; v < 8; ++v)
                dst[(size_t)v * PP] = acc[mi][ni][v];
        }
    }
}

// ---------------------------------------------------------------------------
// Host launcher
// ---------------------------------------------------------------------------
extern "C" void kernel_launch(void* const* d_in, const int* in_sizes, int n_in,
                              void* d_out, int out_size, void* d_ws, size_t ws_size,
                              hipStream_t stream) {
    const float* x        = (const float*)d_in[0];  // [B,C,H,W]
    const float* router_w = (const float*)d_in[1];  // [E,C,1,1]
    const float* router_b = (const float*)d_in[2];  // [E]
    const float* expert_w = (const float*)d_in[3];  // [E,O,C,3,3]
    float* out = (float*)d_out;                     // [B,O,H,W]

    // Workspace layout (~105 MB):
    //   pooled f32 [B,C]          @ 0          (32 KB)
    //   attn   f32 [B,E]          @ 32768
    //   x bf16 [B,C,H,W]          @ 65536      (64 MB)
    //   W_b bf16 [B,O,9,C]        @ 65536+64MB (36 MB, tap-major)
    char* w = (char*)d_ws;
    float*          pooled = (float*)(w);
    float*          attn   = (float*)(w + 32768);
    unsigned short* xbf    = (unsigned short*)(w + 65536);
    unsigned short* wbw    = (unsigned short*)(w + 65536 + (size_t)BB * CC * PP * 2);

    pool_convert_kernel<<<BB * CC, 256, 0, stream>>>(x, pooled, xbf);
    router_kernel<<<1, 32, 0, stream>>>(pooled, router_w, router_b, attn);
    combine_kernel<<<(BB * (long)OO * KK) / 512, 256, 0, stream>>>(
        attn, expert_w, (unsigned int*)wbw);
    condconv_gemm_kernel<<<dim3(PP / 128, OO / 128, BB), 256, 0, stream>>>(
        xbf, wbw, out);
}
